// DeformConv_50577534877940
// MI455X (gfx1250) — matrix operand order repaired and checked by reading, not compiled, not run on hardware
//
#include <hip/hip_runtime.h>

#define BATCH 8
#define CIN   128
#define HH    64
#define WW    64
#define COUT  256
#define HW    4096            // 64*64
#define KDIM  1152            // CIN*9
#define NPIX  32768           // BATCH*HW
#define KSTEP 64
#define NSTEP (KDIM / KSTEP)  // 18

typedef _Float16 h16v __attribute__((ext_vector_type(16)));
typedef _Float16 h8v  __attribute__((ext_vector_type(8)));
typedef float    f8v  __attribute__((ext_vector_type(8)));
typedef unsigned int u32x4 __attribute__((ext_vector_type(4)));
typedef int          i32x8 __attribute__((ext_vector_type(8)));
typedef int          i32x4 __attribute__((ext_vector_type(4)));

#if defined(__has_builtin)
#  if __has_builtin(__builtin_amdgcn_tensor_load_to_lds) && \
      __has_builtin(__builtin_amdgcn_s_wait_tensorcnt)
#    define HAVE_TDM 1
#  endif
#endif
#ifndef HAVE_TDM
#  define HAVE_TDM 0
#endif

// ---------------- workspace layout (bytes) ----------------
#define WS_XT   0ull                       // float [8][64][64][128]   16,777,216
#define WS_OFF  16777216ull                // float [8][18][4096]       2,359,296
#define WS_MOD  19136512ull                // float [8][9][4096]        1,179,648
#define WS_WH   20316160ull                // f16   [256][1152]           589,824
#define WS_COL  20905984ull                // f16   [32768][1152]      75,497,472
#define WS_NEED 96403456ull

// NCHW -> NHWC transpose of x (channel-contiguous for coalesced sampling gathers)
__global__ void k_transpose(const float* __restrict__ x, float* __restrict__ xT) {
  int idx = blockIdx.x * 256 + threadIdx.x;          // 0 .. 8*128*4096-1
  int c    = idx & 127;
  int rest = idx >> 7;                               // b*4096 + y*64 + xw
  int xw   = rest & 63;
  int y    = (rest >> 6) & 63;
  int b    = rest >> 12;
  xT[idx] = x[(((b * CIN + c) * HH + y) * WW) + xw];
}

// conv_w [256][128][3][3] fp32 -> Wh[o][k], k = n2*128 + c, fp16
__global__ void k_wconv(const float* __restrict__ cw, _Float16* __restrict__ Wh) {
  int idx = blockIdx.x * 256 + threadIdx.x;          // 0 .. 256*1152-1
  int o  = idx / KDIM;
  int k  = idx - o * KDIM;
  int n2 = k >> 7;
  int c  = k & 127;
  Wh[idx] = (_Float16)cw[(o * CIN + c) * 9 + n2];
}

// offset conv (f<18, linear) + modulation conv (f>=18, sigmoid); pad=1, stride=1
__global__ void k_offmod(const float* __restrict__ xT,
                         const float* __restrict__ ow, const float* __restrict__ ob,
                         const float* __restrict__ mw, const float* __restrict__ mb,
                         float* __restrict__ OFF, float* __restrict__ MOD) {
  int p = blockIdx.x * 256 + threadIdx.x;            // pixel 0..32767
  int f = blockIdx.y;                                // filter 0..26
  int b = p >> 12, rem = p & 4095, hp = rem >> 6, wp = rem & 63;
  const float* wf; float bias;
  if (f < 18) { wf = ow + f * KDIM;        bias = ob[f];      }
  else        { wf = mw + (f - 18) * KDIM; bias = mb[f - 18]; }
  float acc = bias;
  for (int ky = 0; ky < 3; ++ky) {
    int y = hp + ky - 1;
    if ((unsigned)y >= 64u) continue;
    for (int kx = 0; kx < 3; ++kx) {
      int xw = wp + kx - 1;
      if ((unsigned)xw >= 64u) continue;
      const float* base = xT + (((b * HH + y) * WW) + xw) * CIN;
      int kpos = ky * 3 + kx;
#pragma unroll 8
      for (int c = 0; c < CIN; ++c)
        acc = fmaf(base[c], wf[c * 9 + kpos], acc);
    }
  }
  if (f < 18) OFF[(b * 18 + f) * HW + rem] = acc;
  else        MOD[(b * 9 + (f - 18)) * HW + rem] = 1.f / (1.f + __expf(-acc));
}

// deformable bilinear sampling * modulation -> Col[pix][n2*128+c] (f16)
// one wave handles one (pix, n2); lanes stride over channels
__global__ void k_sample(const float* __restrict__ xT,
                         const float* __restrict__ OFF,
                         const float* __restrict__ MOD,
                         _Float16* __restrict__ Col) {
  int wg   = blockIdx.x * 8 + (threadIdx.x >> 5);    // 0 .. NPIX*9-1
  int lane = threadIdx.x & 31;
  int pix  = wg / 9;
  int n2   = wg - pix * 9;
  int b = pix >> 12, rem = pix & 4095, h_i = rem >> 6, w_i = rem & 63;
  int a = n2 / 3, d = n2 - a * 3;
  // faithful inverse of the reference's reshape/transpose scramble
  int t  = 192 * d + 3 * h_i + a;
  int wp = t / 9;
  int n  = t - wp * 9;
  int hp = w_i;
  float offy = OFF[(b * 18 + n) * HW + hp * 64 + wp];
  float offx = OFF[(b * 18 + 9 + n) * HW + hp * 64 + wp];
  float mval = MOD[(b * 9 + n2) * HW + rem];
  float py = (float)(hp + n / 3) + offy;             // (hp+1) + (n/3 - 1) + off
  float px = (float)(wp + (n - (n / 3) * 3)) + offx; // (wp+1) + (n%3 - 1) + off
  float iy = fminf(fmaxf(py, 0.f), 63.f);
  float ix = fminf(fmaxf(px, 0.f), 63.f);
  float y0f = floorf(iy), x0f = floorf(ix);
  float fy = iy - y0f, fx = ix - x0f;
  int y0 = (int)y0f, x0 = (int)x0f;
  int y1 = y0 + 1; if (y1 > 63) y1 = 63;
  int x1 = x0 + 1; if (x1 > 63) x1 = 63;
  float w00 = (1.f - fy) * (1.f - fx) * mval;
  float w01 = (1.f - fy) * fx * mval;
  float w10 = fy * (1.f - fx) * mval;
  float w11 = fy * fx * mval;
  const float* r0  = xT + (size_t)(b * HH + y0) * WW * CIN;
  const float* r1  = xT + (size_t)(b * HH + y1) * WW * CIN;
  const float* p00 = r0 + x0 * CIN;
  const float* p01 = r0 + x1 * CIN;
  const float* p10 = r1 + x0 * CIN;
  const float* p11 = r1 + x1 * CIN;
  _Float16* dst = Col + (size_t)pix * KDIM + n2 * CIN;
#pragma unroll
  for (int j = 0; j < 4; ++j) {
    int c = lane + 32 * j;
    float v = w00 * p00[c] + w01 * p01[c] + w10 * p10[c] + w11 * p11[c];
    dst[c] = (_Float16)v;
  }
}

#if HAVE_TDM
// Issue a TDM 2D tile load: rows x tile_w f16 elements, row stride `stride`
// elements, packed contiguously into LDS at lds_off. D# bitfields per CDNA5
// ISA 8.3-8.5: data_size=1 (2B), huge tensor dims (no OOB clip), type=2.
__device__ __forceinline__ void tdm_load_tile(unsigned lds_off, const void* gptr,
                                              unsigned tile_w, unsigned rows,
                                              unsigned stride) {
  unsigned long long ga = (unsigned long long)(uintptr_t)gptr;
  u32x4 g0;
  g0[0] = 1u;                                            // count=1, user mode
  g0[1] = lds_off;                                       // lds_addr
  g0[2] = (unsigned)ga;                                  // global_addr[31:0]
  g0[3] = (unsigned)((ga >> 32) & 0x01FFFFFFu) | 0x80000000u; // ga[56:32], type=2
  i32x8 g1;
  g1[0] = 0x00010000;                                    // wg_mask=0, data_size=2B
  g1[1] = 0;                                             // abar=0, tdim0 lo16 (1<<20 -> 0)
  g1[2] = 16;                                            // tdim0 hi16=16, tdim1 lo16=0
  g1[3] = (int)((tile_w << 16) | 16u);                   // tdim1 hi16=16, tile_dim0
  g1[4] = (int)rows;                                     // tile_dim1, tile_dim2=0
  g1[5] = (int)stride;                                   // tdim0_stride lo32
  g1[6] = 0;                                             // stride0 hi16, stride1 lo16
  g1[7] = 0;                                             // stride1 hi32
  i32x4 z4 = {0, 0, 0, 0};
#if __clang_major__ >= 23
  i32x8 z8 = {0, 0, 0, 0, 0, 0, 0, 0};
  __builtin_amdgcn_tensor_load_to_lds(g0, g1, z4, z4, z8, 0);
#else
  __builtin_amdgcn_tensor_load_to_lds(g0, g1, z4, z4, 0);
#endif
}
#endif

// Fallback cooperative staging: 256 threads copy a 128 x KSTEP f16 tile.
__device__ __forceinline__ void stage_tile(_Float16* dst, const _Float16* src, int tid) {
  int row = tid >> 1, c0 = (tid & 1) * 32;               // 32 halfwords per thread
  const uint4* s = (const uint4*)(src + (size_t)row * KDIM + c0);
  uint4* d = (uint4*)(dst + row * KSTEP + c0);
  d[0] = s[0]; d[1] = s[1]; d[2] = s[2]; d[3] = s[3];
}

// GEMM: out[o,pix] = sum_k Wh[o][k] * Col[pix][k]  (f16 WMMA, f32 accumulate)
// block = 8 waves = 128M x 128N; TDM double-buffered LDS tiles of 128 x 64.
__global__ void __launch_bounds__(256) k_gemm(const _Float16* __restrict__ Wh,
                                              const _Float16* __restrict__ Col,
                                              float* __restrict__ out) {
  __shared__ __attribute__((aligned(128))) _Float16 As[2][128 * KSTEP];
  __shared__ __attribute__((aligned(128))) _Float16 Bs[2][128 * KSTEP];

  int tid  = threadIdx.x;
  int wave = tid >> 5, lane = tid & 31;
  int half = lane >> 4, lr = lane & 15;
  int wm = wave & 1, wn = wave >> 1;
  int Mbase = blockIdx.y * 128;
  int Nbase = blockIdx.x * 128;

  const _Float16* Ag = Wh  + (size_t)Mbase * KDIM;
  const _Float16* Bg = Col + (size_t)Nbase * KDIM;

#if HAVE_TDM
  if (wave == 0) {
    tdm_load_tile((unsigned)(uintptr_t)&As[0][0], Ag,         KSTEP, 128, KDIM);
    tdm_load_tile((unsigned)(uintptr_t)&Bs[0][0], Bg,         KSTEP, 128, KDIM);
    tdm_load_tile((unsigned)(uintptr_t)&As[1][0], Ag + KSTEP, KSTEP, 128, KDIM);
    tdm_load_tile((unsigned)(uintptr_t)&Bs[1][0], Bg + KSTEP, KSTEP, 128, KDIM);
  }
#else
  stage_tile(As[0], Ag,         tid);
  stage_tile(Bs[0], Bg,         tid);
  stage_tile(As[1], Ag + KSTEP, tid);
  stage_tile(Bs[1], Bg + KSTEP, tid);
#endif

  f8v acc[4][2] = {};

  for (int s = 0; s < NSTEP; ++s) {
    int buf = s & 1;
#if HAVE_TDM
    if (wave == 0) {                      // TDM ops complete in order per wave
      if (s + 1 < NSTEP) __builtin_amdgcn_s_wait_tensorcnt(2);
      else               __builtin_amdgcn_s_wait_tensorcnt(0);
    }
#endif
    __syncthreads();                      // tile for step s visible to all waves

    const _Float16* Abuf = &As[buf][0];
    const _Float16* Bbuf = &Bs[buf][0];
#pragma unroll
    for (int ks = 0; ks < KSTEP; ks += 32) {
      h16v afr[4];
#pragma unroll
      for (int i = 0; i < 4; ++i) {
        const _Float16* pa = Abuf + (wm * 64 + 16 * i + lr) * KSTEP + ks + 8 * half;
        h8v lo = *(const h8v*)pa;                         // K = ks + 8h + 0..7
        h8v hi = *(const h8v*)(pa + 16);                  // K = ks + 16 + 8h + 0..7
        afr[i] = __builtin_shufflevector(lo, hi, 0,1,2,3,4,5,6,7,8,9,10,11,12,13,14,15);
      }
      h16v bfr[2];
#pragma unroll
      for (int j = 0; j < 2; ++j) {
        const _Float16* pb = Bbuf + (wn * 32 + 16 * j + lr) * KSTEP + ks + 8 * half;
        h8v blo = *(const h8v*)pb;
        h8v bhi = *(const h8v*)(pb + 16);
        bfr[j] = __builtin_shufflevector(blo, bhi, 0,1,2,3,4,5,6,7,8,9,10,11,12,13,14,15);
      }
#pragma unroll
      for (int i = 0; i < 4; ++i)
#pragma unroll
        for (int j = 0; j < 2; ++j)
          acc[i][j] = __builtin_amdgcn_wmma_f32_16x16x32_f16(
              false, afr[i], false, bfr[j], (short)0, acc[i][j], false, false);
    }
    __syncthreads();                      // all waves done reading buf

    if (s + 2 < NSTEP) {                  // prefetch step s+2 into freed buffer
#if HAVE_TDM
      if (wave == 0) {
        tdm_load_tile((unsigned)(uintptr_t)&As[buf][0], Ag + (s + 2) * KSTEP, KSTEP, 128, KDIM);
        tdm_load_tile((unsigned)(uintptr_t)&Bs[buf][0], Bg + (s + 2) * KSTEP, KSTEP, 128, KDIM);
      }
#else
      stage_tile(As[buf], Ag + (s + 2) * KSTEP, tid);
      stage_tile(Bs[buf], Bg + (s + 2) * KSTEP, tid);
#endif
    }
  }

  int b = (blockIdx.x * 128) >> 12;       // uniform: 4096 % 128 == 0
#pragma unroll
  for (int i = 0; i < 4; ++i) {
#pragma unroll
    for (int j = 0; j < 2; ++j) {
      int pix = (Nbase + wn * 32 + 16 * j + lr) & 4095;
#pragma unroll
      for (int r = 0; r < 8; ++r) {
        int o = Mbase + wm * 64 + 16 * i + r + 8 * half;  // D: VGPR r -> row r+8*(lane/16)
        out[((size_t)(b * COUT + o)) * HW + pix] = acc[i][j][r];
      }
    }
  }
}

extern "C" void kernel_launch(void* const* d_in, const int* in_sizes, int n_in,
                              void* d_out, int out_size, void* d_ws, size_t ws_size,
                              hipStream_t stream) {
  if (ws_size < WS_NEED) return;
  const float* x   = (const float*)d_in[0];
  const float* ow  = (const float*)d_in[1];
  const float* ob  = (const float*)d_in[2];
  const float* mw  = (const float*)d_in[3];
  const float* mb  = (const float*)d_in[4];
  const float* cw  = (const float*)d_in[5];
  float* out = (float*)d_out;

  char* ws = (char*)d_ws;
  float*    xT  = (float*)(ws + WS_XT);
  float*    OFF = (float*)(ws + WS_OFF);
  float*    MOD = (float*)(ws + WS_MOD);
  _Float16* Wh  = (_Float16*)(ws + WS_WH);
  _Float16* Col = (_Float16*)(ws + WS_COL);

  k_transpose<<<dim3((BATCH * CIN * HW) / 256), 256, 0, stream>>>(x, xT);
  k_wconv<<<dim3((COUT * KDIM) / 256), 256, 0, stream>>>(cw, Wh);
  k_offmod<<<dim3(NPIX / 256, 27), 256, 0, stream>>>(xT, ow, ob, mw, mb, OFF, MOD);
  k_sample<<<dim3((NPIX * 9) / 8), 256, 0, stream>>>(xT, OFF, MOD, Col);
  k_gemm<<<dim3(NPIX / 128, COUT / 128), 256, 0, stream>>>(Wh, Col, out);
}
